// brain_41695542510064
// MI455X (gfx1250) — matrix operand order repaired and checked by reading, not compiled, not run on hardware
//
#include <hip/hip_runtime.h>
#include <cstdint>
#include <cstddef>

// ---------------------------------------------------------------------------
// MI455X (gfx1250, wave32) fused causal MHA:
//   prep(convert/transpose bf16) -> QKV GEMM (WMMA bf16) ->
//   flash attention (WMMA bf16, online softmax) -> proj GEMM (WMMA bf16)
// ---------------------------------------------------------------------------

constexpr int kBatch = 4;
constexpr int kSeq   = 2048;
constexpr int kEmbd  = 1024;
constexpr int kHeads = 16;
constexpr int kHeadD = 64;
constexpr int kRows  = kBatch * kSeq;   // 8192
constexpr int kQKVN  = 3 * kEmbd;       // 3072

typedef __attribute__((ext_vector_type(16))) __bf16 bf16x16;
typedef __attribute__((ext_vector_type(8)))  __bf16 bf16x8;
typedef __attribute__((ext_vector_type(8)))  float  f32x8;
typedef __attribute__((ext_vector_type(4)))  int    v4i;

union BF16Frag { bf16x16 v; bf16x8 h[2]; };

#if __has_builtin(__builtin_amdgcn_global_load_async_to_lds_b128) && \
    __has_builtin(__builtin_amdgcn_s_wait_asynccnt)
#define HAVE_ASYNC 1
typedef __attribute__((address_space(1))) v4i GV4;
typedef __attribute__((address_space(3))) v4i LV4;
__device__ __forceinline__ void async_copy16(const void* g, void* l) {
  __builtin_amdgcn_global_load_async_to_lds_b128((GV4*)g, (LV4*)l, 0, 0);
}
#else
#define HAVE_ASYNC 0
#endif

__device__ __forceinline__ f32x8 zero_f32x8() {
  f32x8 z = {0.f, 0.f, 0.f, 0.f, 0.f, 0.f, 0.f, 0.f};
  return z;
}

__device__ __forceinline__ f32x8 wmma_bf16(bf16x16 a, bf16x16 b, f32x8 c) {
  // v_wmma_f32_16x16x32_bf16
  return __builtin_amdgcn_wmma_f32_16x16x32_bf16(
      /*neg_a=*/false, a, /*neg_b=*/false, b,
      /*c_mod=*/(short)0, c, /*reuse_a=*/false, /*reuse_b=*/false);
}

// ---------------------------------------------------------------------------
// Prep kernels: f32 -> bf16 convert (and transpose for weights)
// ---------------------------------------------------------------------------
__global__ __launch_bounds__(256) void cvt_kernel(const float* __restrict__ in,
                                                  __bf16* __restrict__ out, int n) {
  int i = blockIdx.x * 256 + threadIdx.x;
  if (i < n) out[i] = (__bf16)in[i];
}

// in: [K][N] row-major (f32) -> outT: [N][K] row-major (bf16)
__global__ __launch_bounds__(256) void cvt_transpose_kernel(const float* __restrict__ in,
                                                            __bf16* __restrict__ outT,
                                                            int K, int N) {
  int i = blockIdx.x * 256 + threadIdx.x;
  if (i < K * N) {
    int n  = i / K;
    int kk = i - n * K;
    outT[i] = (__bf16)in[(size_t)kk * N + n];
  }
}

// ---------------------------------------------------------------------------
// GEMM: C[M,N] = A[M,K] * B[K,N] + bias, A row-major bf16, BT = B^T [N][K] bf16.
// Block tile 128x128, 8 waves (4M x 2N), wave tile 32x64 = 2x4 WMMA accums.
// QKV==1: split columns into q/k/vT bf16 outputs; QKV==0: f32 output.
// ---------------------------------------------------------------------------
template <int QKV>
__global__ __launch_bounds__(256) void gemm_kernel(
    const __bf16* __restrict__ A, const __bf16* __restrict__ BT,
    const float* __restrict__ bias, float* __restrict__ Cout,
    __bf16* __restrict__ qo, __bf16* __restrict__ ko, __bf16* __restrict__ vTo,
    int Ntot, int K) {
  __shared__ __align__(16) __bf16 As[128][32];
  __shared__ __align__(16) __bf16 Bs[128][32];

  const int tid    = threadIdx.x;
  const int wave   = tid >> 5;
  const int lane   = tid & 31;
  const int lane16 = lane & 15;
  const int khalf  = lane >> 4;
  const int wm     = wave >> 1;   // 0..3
  const int wn     = wave & 1;    // 0..1
  const int m0     = blockIdx.y * 128;
  const int n0     = blockIdx.x * 128;

  f32x8 acc[2][4];
#pragma unroll
  for (int i = 0; i < 2; i++)
#pragma unroll
    for (int j = 0; j < 4; j++) acc[i][j] = zero_f32x8();

  for (int k0 = 0; k0 < K; k0 += 32) {
    __syncthreads();  // previous iteration's LDS readers done
#pragma unroll
    for (int c = 0; c < 2; c++) {
      int li  = (tid + c * 256) * 8;  // element index in 128x32 tile
      int row = li >> 5;
      int kk  = li & 31;
      const __bf16* gA = A + (size_t)(m0 + row) * K + k0 + kk;
      const __bf16* gB = BT + (size_t)(n0 + row) * K + k0 + kk;
#if HAVE_ASYNC
      async_copy16(gA, &As[row][kk]);
      async_copy16(gB, &Bs[row][kk]);
#else
      *(bf16x8*)&As[row][kk] = *(const bf16x8*)gA;
      *(bf16x8*)&Bs[row][kk] = *(const bf16x8*)gB;
#endif
    }
#if HAVE_ASYNC
    __builtin_amdgcn_s_wait_asynccnt(0);
#endif
    __syncthreads();

    // A fragments (16-bit A 16x32 interleave: e<8 -> K=khalf*8+e ; e>=8 -> +16)
    BF16Frag af[2];
#pragma unroll
    for (int m2 = 0; m2 < 2; m2++) {
      const __bf16* ap = &As[wm * 32 + m2 * 16 + lane16][khalf * 8];
      af[m2].h[0] = *(const bf16x8*)ap;
      af[m2].h[1] = *(const bf16x8*)(ap + 16);
    }
    // B fragments (16-bit B 32x16: lane=N, e<8 -> K=khalf*8+e ; e>=8 -> +16)
    BF16Frag bfr[4];
#pragma unroll
    for (int j = 0; j < 4; j++) {
      const __bf16* bp = &Bs[wn * 64 + j * 16 + lane16][khalf * 8];
      bfr[j].h[0] = *(const bf16x8*)bp;
      bfr[j].h[1] = *(const bf16x8*)(bp + 16);
    }
#pragma unroll
    for (int m2 = 0; m2 < 2; m2++)
#pragma unroll
      for (int j = 0; j < 4; j++)
        acc[m2][j] = wmma_bf16(af[m2].v, bfr[j].v, acc[m2][j]);
  }

  // Store. C layout: element r -> row M = r + 8*khalf, col N = lane16.
#pragma unroll
  for (int m2 = 0; m2 < 2; m2++) {
#pragma unroll
    for (int j = 0; j < 4; j++) {
#pragma unroll
      for (int r = 0; r < 8; r++) {
        int row   = m0 + wm * 32 + m2 * 16 + r + 8 * khalf;
        int col   = n0 + wn * 64 + j * 16 + lane16;
        float val = acc[m2][j][r] + bias[col];
        if (QKV) {
          int sec = col >> 10, cc = col & 1023;
          int h = cc >> 6, d = cc & 63;
          int b = row >> 11, t = row & 2047;
          int bh = b * kHeads + h;
          if (sec == 0)
            qo[((size_t)bh * kSeq + t) * kHeadD + d] = (__bf16)val;
          else if (sec == 1)
            ko[((size_t)bh * kSeq + t) * kHeadD + d] = (__bf16)val;
          else  // v stored transposed: [B,H,D,T]
            vTo[((size_t)bh * kHeadD + d) * kSeq + t] = (__bf16)val;
        } else {
          Cout[(size_t)row * Ntot + col] = val;
        }
      }
    }
  }
}

// ---------------------------------------------------------------------------
// Flash attention (causal). Grid: (B*H) * (T/128). 4 waves x 32 query rows.
// q,k: [B,H,T,D] bf16 ; vT: [B,H,D,T] bf16 ; ybf out: [B*T, C] bf16.
// ---------------------------------------------------------------------------
__global__ __launch_bounds__(128) void attn_kernel(const __bf16* __restrict__ q,
                                                   const __bf16* __restrict__ k,
                                                   const __bf16* __restrict__ vT,
                                                   __bf16* __restrict__ ybf) {
  __shared__ __align__(16) __bf16 Ps[4][32][64];  // per-wave P tile (bf16)

  const int tid    = threadIdx.x;
  const int wave   = tid >> 5;
  const int lane   = tid & 31;
  const int lane16 = lane & 15;
  const int khalf  = lane >> 4;
  const int bh     = blockIdx.x >> 4;  // 16 q-blocks per (b,h)
  const int qblk   = blockIdx.x & 15;
  const int b      = bh >> 4;
  const int h      = bh & 15;
  const int qr0    = qblk * 128 + wave * 32;

  const __bf16* qh = q + (size_t)bh * kSeq * kHeadD;
  const __bf16* kh = k + (size_t)bh * kSeq * kHeadD;
  const __bf16* vh = vT + (size_t)bh * kHeadD * kSeq;

  // Preload Q fragments: 2 M-tiles x 2 K-steps over D=64.
  BF16Frag qf[2][2];
#pragma unroll
  for (int m2 = 0; m2 < 2; m2++)
#pragma unroll
    for (int ks = 0; ks < 2; ks++) {
      const __bf16* p = qh + (size_t)(qr0 + m2 * 16 + lane16) * kHeadD + ks * 32 + khalf * 8;
      qf[m2][ks].h[0] = *(const bf16x8*)p;
      qf[m2][ks].h[1] = *(const bf16x8*)(p + 16);
    }

  f32x8 o[2][4];
  float mrow[2][8], lrow[2][8];
#pragma unroll
  for (int m2 = 0; m2 < 2; m2++) {
#pragma unroll
    for (int nd = 0; nd < 4; nd++) o[m2][nd] = zero_f32x8();
#pragma unroll
    for (int r = 0; r < 8; r++) { mrow[m2][r] = -1e30f; lrow[m2][r] = 0.f; }
  }

  const int nkb = 2 * qblk + 2;  // causal: key blocks of 64 up to q block end
  for (int kbi = 0; kbi < nkb; kbi++) {
    const int kb = kbi * 64;

    // S = Q @ K^T (contraction over D); K fragments loaded just-in-time.
    f32x8 s_[2][4];
#pragma unroll
    for (int j = 0; j < 4; j++) {
      BF16Frag kf0, kf1;
      {
        const __bf16* p0 = kh + (size_t)(kb + j * 16 + lane16) * kHeadD + khalf * 8;
        kf0.h[0] = *(const bf16x8*)p0;
        kf0.h[1] = *(const bf16x8*)(p0 + 16);
        const __bf16* p1 = p0 + 32;
        kf1.h[0] = *(const bf16x8*)p1;
        kf1.h[1] = *(const bf16x8*)(p1 + 16);
      }
#pragma unroll
      for (int m2 = 0; m2 < 2; m2++) {
        f32x8 t = wmma_bf16(qf[m2][0].v, kf0.v, zero_f32x8());
        s_[m2][j] = wmma_bf16(qf[m2][1].v, kf1.v, t);
      }
    }

    // Scale + causal mask.
#pragma unroll
    for (int m2 = 0; m2 < 2; m2++)
#pragma unroll
      for (int j = 0; j < 4; j++)
#pragma unroll
        for (int r = 0; r < 8; r++) {
          int keyg = kb + j * 16 + lane16;
          int qg   = qr0 + m2 * 16 + r + 8 * khalf;
          float sv = s_[m2][j][r] * 0.125f;  // 1/sqrt(64)
          s_[m2][j][r] = (keyg <= qg) ? sv : -1e30f;
        }

    // Online softmax. Row M=r+8*khalf lives in one 16-lane half: xor masks 1..8.
#pragma unroll
    for (int m2 = 0; m2 < 2; m2++) {
#pragma unroll
      for (int r = 0; r < 8; r++) {
        float mx = s_[m2][0][r];
#pragma unroll
        for (int j = 1; j < 4; j++) mx = fmaxf(mx, s_[m2][j][r]);
#pragma unroll
        for (int off = 1; off < 16; off <<= 1) mx = fmaxf(mx, __shfl_xor(mx, off, 32));
        float mnew  = fmaxf(mrow[m2][r], mx);
        float alpha = __expf(mrow[m2][r] - mnew);
        mrow[m2][r] = mnew;
        float rs = 0.f;
#pragma unroll
        for (int j = 0; j < 4; j++) {
          float pv = __expf(s_[m2][j][r] - mnew);
          rs += pv;
          Ps[wave][m2 * 16 + r + 8 * khalf][j * 16 + lane16] = (__bf16)pv;
        }
#pragma unroll
        for (int off = 1; off < 16; off <<= 1) rs += __shfl_xor(rs, off, 32);
        lrow[m2][r] = lrow[m2][r] * alpha + rs;
#pragma unroll
        for (int nd = 0; nd < 4; nd++) o[m2][nd][r] *= alpha;
      }
    }

    __syncthreads();  // P tile in LDS ready (all waves same trip count)

    // P fragments (A layout) from LDS.
    BF16Frag pf[2][2];
#pragma unroll
    for (int m2 = 0; m2 < 2; m2++)
#pragma unroll
      for (int ks = 0; ks < 2; ks++) {
        const __bf16* pp = &Ps[wave][m2 * 16 + lane16][ks * 32 + khalf * 8];
        pf[m2][ks].h[0] = *(const bf16x8*)pp;
        pf[m2][ks].h[1] = *(const bf16x8*)(pp + 16);
      }

    // O += P @ V ; V fragments from vT (contiguous keys per output dim).
#pragma unroll
    for (int nd = 0; nd < 4; nd++) {
      BF16Frag vf0, vf1;
      {
        const __bf16* p0 = vh + (size_t)(nd * 16 + lane16) * kSeq + kb + khalf * 8;
        vf0.h[0] = *(const bf16x8*)p0;
        vf0.h[1] = *(const bf16x8*)(p0 + 16);
        const __bf16* p1 = p0 + 32;
        vf1.h[0] = *(const bf16x8*)p1;
        vf1.h[1] = *(const bf16x8*)(p1 + 16);
      }
#pragma unroll
      for (int m2 = 0; m2 < 2; m2++) {
        o[m2][nd] = wmma_bf16(pf[m2][0].v, vf0.v, o[m2][nd]);
        o[m2][nd] = wmma_bf16(pf[m2][1].v, vf1.v, o[m2][nd]);
      }
    }
  }

  // Epilogue: normalize and write y (heads re-merged) as bf16 [B*T, C].
#pragma unroll
  for (int m2 = 0; m2 < 2; m2++)
#pragma unroll
    for (int nd = 0; nd < 4; nd++)
#pragma unroll
      for (int r = 0; r < 8; r++) {
        float val = o[m2][nd][r] / lrow[m2][r];
        int t   = qr0 + m2 * 16 + r + 8 * khalf;
        int col = h * kHeadD + nd * 16 + lane16;
        ybf[((size_t)(b * kSeq + t)) * kEmbd + col] = (__bf16)val;
      }
}

// ---------------------------------------------------------------------------
// Host side
// ---------------------------------------------------------------------------
extern "C" void kernel_launch(void* const* d_in, const int* in_sizes, int n_in,
                              void* d_out, int out_size, void* d_ws, size_t ws_size,
                              hipStream_t stream) {
  (void)in_sizes; (void)n_in; (void)out_size; (void)ws_size;
  const float* x      = (const float*)d_in[0];  // [4,2048,1024]
  const float* W_attn = (const float*)d_in[1];  // [1024,3072]
  const float* b_attn = (const float*)d_in[2];  // [3072]
  const float* W_proj = (const float*)d_in[3];  // [1024,1024]
  const float* b_proj = (const float*)d_in[4];  // [1024]
  float* out = (float*)d_out;                   // [4,2048,1024] f32

  char* ws = (char*)d_ws;
  const size_t szX     = (size_t)kRows * kEmbd * 2;   // 16 MB
  const size_t szWa    = (size_t)kQKVN * kEmbd * 2;   // 6 MB
  const size_t szWp    = (size_t)kEmbd * kEmbd * 2;   // 2 MB
  const size_t szHead  = (size_t)kBatch * kHeads * kSeq * kHeadD * 2;  // 16 MB

  __bf16* xbf    = (__bf16*)(ws);
  __bf16* WattnT = (__bf16*)(ws + szX);
  __bf16* WprojT = (__bf16*)(ws + szX + szWa);
  __bf16* qb     = (__bf16*)(ws + szX + szWa + szWp);
  __bf16* kb     = (__bf16*)(ws + szX + szWa + szWp + szHead);
  __bf16* vTb    = (__bf16*)(ws + szX + szWa + szWp + 2 * szHead);
  __bf16* ybf    = (__bf16*)(ws + szX + szWa + szWp + 3 * szHead);

  cvt_kernel<<<(kRows * kEmbd) / 256, 256, 0, stream>>>(x, xbf, kRows * kEmbd);
  cvt_transpose_kernel<<<(kEmbd * kQKVN) / 256, 256, 0, stream>>>(W_attn, WattnT, kEmbd, kQKVN);
  cvt_transpose_kernel<<<(kEmbd * kEmbd) / 256, 256, 0, stream>>>(W_proj, WprojT, kEmbd, kEmbd);

  gemm_kernel<1><<<dim3(kQKVN / 128, kRows / 128), 256, 0, stream>>>(
      xbf, WattnT, b_attn, nullptr, qb, kb, vTb, kQKVN, kEmbd);

  attn_kernel<<<dim3(kBatch * kHeads * (kSeq / 128)), 128, 0, stream>>>(qb, kb, vTb, ybf);

  gemm_kernel<0><<<dim3(kEmbd / 128, kRows / 128), 256, 0, stream>>>(
      ybf, WprojT, b_proj, out, nullptr, nullptr, nullptr, kEmbd, kEmbd);
}
